// MultiHeadAttention_7550552506995
// MI455X (gfx1250) — matrix operand order repaired and checked by reading, not compiled, not run on hardware
//
#include <hip/hip_runtime.h>
#include <hip/hip_bf16.h>

// Problem dims (fixed by reference)
#define BB   2
#define SS   2048
#define DD   1024
#define HH   16
#define DKK  64
#define TOPKN 64
#define NEGV (-10000.0f)
#define LDP  2052   // LDS row pitch (floats) for 2048 cols + pad
#define VSTAGE_BYTES 8192          // 2 x (32x64 bf16) staging buffers
#define ATTN_LDS (VSTAGE_BYTES + 32 * LDP * 4)

typedef __attribute__((ext_vector_type(16))) __bf16         v16bf;
typedef __attribute__((ext_vector_type(8)))  float          v8f;
typedef __attribute__((ext_vector_type(8)))  unsigned short v8us;
typedef __attribute__((ext_vector_type(4)))  unsigned int   u32x4;
typedef __attribute__((ext_vector_type(8)))  int            i32x8;
typedef __attribute__((ext_vector_type(4)))  int            i32x4;

union Frag16 {
  v16bf bf;
  v8us  us2[2];
  unsigned short us[16];
};

__device__ __forceinline__ unsigned short f2bf(float f) {
  union { float f; unsigned u; } v; v.f = f;
  unsigned r = v.u + 0x7FFFu + ((v.u >> 16) & 1u);  // RNE truncate
  return (unsigned short)(r >> 16);
}

// ---------------- Tensor Data Mover availability ----------------
#if defined(__has_builtin)
#  if __has_builtin(__builtin_amdgcn_tensor_load_to_lds) && \
      __has_builtin(__builtin_amdgcn_s_wait_tensorcnt)
#    define HAVE_TDM 1
#  endif
#endif
#ifndef HAVE_TDM
#  define HAVE_TDM 0
#endif

#if HAVE_TDM
// Issue a TDM load of a 32-row x 64-col bf16 tile (contiguous, stride 64)
// from global memory into LDS at byte offset lds_off. D# packing per ISA
// 08_async_tensor §8.3/8.4 (group0: count/lds/global/type, group1: dims).
__device__ __forceinline__ void tdm_load_tile(unsigned lds_off,
                                              const unsigned short* gsrc) {
  unsigned long long ga = (unsigned long long)(size_t)gsrc;
  u32x4 g0 = {
    1u,                                    // [1:0] count=1 (valid), user mode
    lds_off,                               // [63:32] lds_addr (bytes)
    (unsigned)(ga & 0xFFFFFFFFu),          // [95:64] global_addr lo
    (unsigned)((ga >> 32) & 0x01FFFFFFu)   // [120:96] global_addr hi
      | 0x80000000u                        // [127:126] type = 2 ("image")
  };
  i32x8 g1 = {
    (int)(1u << 16),        // [17:16] data_size=1 -> 2 bytes; wg_mask=0
    (int)(64u << 16),       // [79:48] tensor_dim0=64 (lo16); atomic_bar=0
    (int)(32u << 16),       // [111:80] tensor_dim1=32 (lo16); dim0 hi=0
    (int)(64u << 16),       // [127:112] tile_dim0=64; dim1 hi=0
    (int)(32u),             // [143:128] tile_dim1=32; tile_dim2=0
    64,                     // [207:160] tensor_dim0_stride=64 (lo32)
    0,                      // stride hi / dim1_stride lo
    0
  };
  i32x4 z4 = {0, 0, 0, 0};
#if defined(__clang_major__) && __clang_major__ >= 23
  i32x8 z8 = {0, 0, 0, 0, 0, 0, 0, 0};
  __builtin_amdgcn_tensor_load_to_lds(g0, g1, z4, z4, z8, 0);
#else
  __builtin_amdgcn_tensor_load_to_lds(g0, g1, z4, z4, 0);
#endif
}
#endif

// ---------------------------------------------------------------- convert
__global__ void cvt_f32_bf16(const float* __restrict__ x,
                             unsigned short* __restrict__ y, int n) {
  for (int i = blockIdx.x * blockDim.x + threadIdx.x; i < n;
       i += gridDim.x * blockDim.x)
    y[i] = f2bf(x[i]);
}

// ---------------------------------------------------------------- projection GEMM
// Out = X(bf16,[M=B*S][K=D]) @ W^T(bf16,[N=D][K=D]) + bias, head-major out.
// Each wave computes a 16x64 strip: one A fragment feeds 4 wmma per k-step.
__global__ void proj_gemm(const unsigned short* __restrict__ X,
                          const unsigned short* __restrict__ W,
                          const float* __restrict__ bias,
                          unsigned short* __restrict__ Out) {
  const int K = DD;
  int wid  = blockIdx.x * (blockDim.x >> 5) + (threadIdx.x >> 5);
  int nt4  = wid & 15;          // N/64 = 16 strips
  int mt   = wid >> 4;          // M/16 = 256
  int l    = threadIdx.x & 31, lo16 = l & 15, hi = l >> 4;

  const unsigned short* aRow = X + (size_t)(mt * 16 + lo16) * K;
  const unsigned short* bRow0 = W + (size_t)(nt4 * 64 +  0 + lo16) * K;
  const unsigned short* bRow1 = W + (size_t)(nt4 * 64 + 16 + lo16) * K;
  const unsigned short* bRow2 = W + (size_t)(nt4 * 64 + 32 + lo16) * K;
  const unsigned short* bRow3 = W + (size_t)(nt4 * 64 + 48 + lo16) * K;

  v8f c0 = {}, c1 = {}, c2 = {}, c3 = {};
  for (int k0 = 0; k0 < K; k0 += 32) {
    __builtin_prefetch(aRow + k0 + 128, 0, 0);
    Frag16 fa, fb;
    fa.us2[0] = *(const v8us*)(aRow + k0 + hi * 8);
    fa.us2[1] = *(const v8us*)(aRow + k0 + hi * 8 + 16);

    fb.us2[0] = *(const v8us*)(bRow0 + k0 + hi * 8);
    fb.us2[1] = *(const v8us*)(bRow0 + k0 + hi * 8 + 16);
    c0 = __builtin_amdgcn_wmma_f32_16x16x32_bf16(false, fa.bf, false, fb.bf,
                                                 (short)0, c0, false, false);
    fb.us2[0] = *(const v8us*)(bRow1 + k0 + hi * 8);
    fb.us2[1] = *(const v8us*)(bRow1 + k0 + hi * 8 + 16);
    c1 = __builtin_amdgcn_wmma_f32_16x16x32_bf16(false, fa.bf, false, fb.bf,
                                                 (short)0, c1, false, false);
    fb.us2[0] = *(const v8us*)(bRow2 + k0 + hi * 8);
    fb.us2[1] = *(const v8us*)(bRow2 + k0 + hi * 8 + 16);
    c2 = __builtin_amdgcn_wmma_f32_16x16x32_bf16(false, fa.bf, false, fb.bf,
                                                 (short)0, c2, false, false);
    fb.us2[0] = *(const v8us*)(bRow3 + k0 + hi * 8);
    fb.us2[1] = *(const v8us*)(bRow3 + k0 + hi * 8 + 16);
    c3 = __builtin_amdgcn_wmma_f32_16x16x32_bf16(false, fa.bf, false, fb.bf,
                                                 (short)0, c3, false, false);
  }

  v8f acc[4] = {c0, c1, c2, c3};
  for (int t = 0; t < 4; ++t) {
    int n  = nt4 * 64 + t * 16 + lo16;
    float bn = bias[n];
    int h = n >> 6, dk = n & 63;
    for (int r = 0; r < 8; ++r) {
      int m  = mt * 16 + r + hi * 8;
      int b_ = m >> 11, s = m & (SS - 1);
      Out[(((size_t)b_ * HH + h) * SS + s) * DKK + dk] = f2bf(acc[t][r] + bn);
    }
  }
}

// ---------------------------------------------------------------- output GEMM
__global__ void out_gemm(const unsigned short* __restrict__ X,
                         const unsigned short* __restrict__ W,
                         const float* __restrict__ bias,
                         float* __restrict__ Out) {
  const int K = DD;
  int wid  = blockIdx.x * (blockDim.x >> 5) + (threadIdx.x >> 5);
  int nt4  = wid & 15;
  int mt   = wid >> 4;
  int l    = threadIdx.x & 31, lo16 = l & 15, hi = l >> 4;

  const unsigned short* aRow = X + (size_t)(mt * 16 + lo16) * K;
  const unsigned short* bRow0 = W + (size_t)(nt4 * 64 +  0 + lo16) * K;
  const unsigned short* bRow1 = W + (size_t)(nt4 * 64 + 16 + lo16) * K;
  const unsigned short* bRow2 = W + (size_t)(nt4 * 64 + 32 + lo16) * K;
  const unsigned short* bRow3 = W + (size_t)(nt4 * 64 + 48 + lo16) * K;

  v8f c0 = {}, c1 = {}, c2 = {}, c3 = {};
  for (int k0 = 0; k0 < K; k0 += 32) {
    __builtin_prefetch(aRow + k0 + 128, 0, 0);
    Frag16 fa, fb;
    fa.us2[0] = *(const v8us*)(aRow + k0 + hi * 8);
    fa.us2[1] = *(const v8us*)(aRow + k0 + hi * 8 + 16);

    fb.us2[0] = *(const v8us*)(bRow0 + k0 + hi * 8);
    fb.us2[1] = *(const v8us*)(bRow0 + k0 + hi * 8 + 16);
    c0 = __builtin_amdgcn_wmma_f32_16x16x32_bf16(false, fa.bf, false, fb.bf,
                                                 (short)0, c0, false, false);
    fb.us2[0] = *(const v8us*)(bRow1 + k0 + hi * 8);
    fb.us2[1] = *(const v8us*)(bRow1 + k0 + hi * 8 + 16);
    c1 = __builtin_amdgcn_wmma_f32_16x16x32_bf16(false, fa.bf, false, fb.bf,
                                                 (short)0, c1, false, false);
    fb.us2[0] = *(const v8us*)(bRow2 + k0 + hi * 8);
    fb.us2[1] = *(const v8us*)(bRow2 + k0 + hi * 8 + 16);
    c2 = __builtin_amdgcn_wmma_f32_16x16x32_bf16(false, fa.bf, false, fb.bf,
                                                 (short)0, c2, false, false);
    fb.us2[0] = *(const v8us*)(bRow3 + k0 + hi * 8);
    fb.us2[1] = *(const v8us*)(bRow3 + k0 + hi * 8 + 16);
    c3 = __builtin_amdgcn_wmma_f32_16x16x32_bf16(false, fa.bf, false, fb.bf,
                                                 (short)0, c3, false, false);
  }

  v8f acc[4] = {c0, c1, c2, c3};
  for (int t = 0; t < 4; ++t) {
    int n = nt4 * 64 + t * 16 + lo16;
    float bn = bias[n];
    for (int r = 0; r < 8; ++r) {
      int m = mt * 16 + r + hi * 8;
      Out[(size_t)m * DD + n] = acc[t][r] + bn;
    }
  }
}

// ---------------------------------------------------------------- attention
// One workgroup (8 waves / 256 thr) per (b,h, 32-row query block).
// Dynamic LDS: [ V double-buffer 8KB | float sc[32][LDP] 256.5KB ].
__global__ void topk_attn(const unsigned short* __restrict__ Qp,
                          const unsigned short* __restrict__ Kp,
                          const unsigned short* __restrict__ Vp,
                          unsigned short* __restrict__ AO) {
  extern __shared__ char smem[];
  unsigned short* vls = (unsigned short*)smem;              // 2 x 2048 bf16
  float* sc = (float*)(smem + VSTAGE_BYTES);                // [32][LDP]

  int blk = blockIdx.x;
  int bh  = blk >> 6;            // 64 row-blocks per (b,h)
  int rb  = blk & 63;
  int b   = bh >> 4, h = bh & 15;
  int qs  = rb * 32;

  const unsigned short* Qh = Qp + (size_t)bh * SS * DKK;
  const unsigned short* Kh = Kp + (size_t)bh * SS * DKK;
  const unsigned short* Vh = Vp + (size_t)bh * SS * DKK;

  int tid = threadIdx.x, w = tid >> 5, l = tid & 31;
  int lo16 = l & 15, hi = l >> 4;

  // ---- Step 1: S = (Q Kt)/8 with causal mask, into LDS.
  // Waves 0-3 handle rt=0, waves 4-7 handle rt=1; Q fragments hoisted so one
  // A-pair feeds 64 wmma.
  {
    int rt = w >> 2, cg = w & 3;
    const unsigned short* aRow = Qh + (size_t)(qs + rt * 16 + lo16) * DKK;
    Frag16 fa0, fa1;
    fa0.us2[0] = *(const v8us*)(aRow + hi * 8);
    fa0.us2[1] = *(const v8us*)(aRow + hi * 8 + 16);
    fa1.us2[0] = *(const v8us*)(aRow + 32 + hi * 8);
    fa1.us2[1] = *(const v8us*)(aRow + 32 + hi * 8 + 16);

    int rowMaxG = qs + rt * 16 + 15;
    for (int ct = cg * 32; ct < cg * 32 + 32; ++ct) {
      if (ct * 16 > rowMaxG) {          // tile entirely above diagonal
        for (int r = 0; r < 8; ++r)
          sc[(rt * 16 + r + hi * 8) * LDP + ct * 16 + lo16] = NEGV;
        continue;
      }
      const unsigned short* bRow = Kh + (size_t)(ct * 16 + lo16) * DKK;
      Frag16 fb;
      v8f c = {};
      fb.us2[0] = *(const v8us*)(bRow + hi * 8);
      fb.us2[1] = *(const v8us*)(bRow + hi * 8 + 16);
      c = __builtin_amdgcn_wmma_f32_16x16x32_bf16(false, fa0.bf, false, fb.bf,
                                                  (short)0, c, false, false);
      fb.us2[0] = *(const v8us*)(bRow + 32 + hi * 8);
      fb.us2[1] = *(const v8us*)(bRow + 32 + hi * 8 + 16);
      c = __builtin_amdgcn_wmma_f32_16x16x32_bf16(false, fa1.bf, false, fb.bf,
                                                  (short)0, c, false, false);
      int jc = ct * 16 + lo16;
      for (int r = 0; r < 8; ++r) {
        int mrow = rt * 16 + r + hi * 8;
        float v  = c[r] * 0.125f;            // 1/sqrt(64)
        if (jc > qs + mrow) v = NEGV;
        sc[mrow * LDP + jc] = v;
      }
    }
  }
  __syncthreads();

  // ---- Steps 2/3: per-row top-64 threshold (bisection) + softmax in place
  for (int rr = 0; rr < 4; ++rr) {
    int i = w + rr * 8;                    // each wave owns 4 rows
    float* row = sc + (size_t)i * LDP;
    int valid = qs + i + 1;

    float mx = -3.0e38f;
    for (int j = l; j < SS; j += 32) mx = fmaxf(mx, row[j]);
    for (int msk = 16; msk; msk >>= 1) mx = fmaxf(mx, __shfl_xor(mx, msk, 32));

    float thr = -3.0e38f;
    if (valid > TOPKN) {
      float lo_ = NEGV - 1.0f, hi_ = mx;
      for (int it = 0; it < 32; ++it) {
        float mid = 0.5f * (lo_ + hi_);
        int cnt = 0;
        for (int j = l; j < SS; j += 32) cnt += (row[j] >= mid) ? 1 : 0;
        for (int msk = 16; msk; msk >>= 1) cnt += __shfl_xor(cnt, msk, 32);
        if (cnt >= TOPKN) lo_ = mid; else hi_ = mid;
      }
      thr = lo_;   // keeps >= 64 elements; ~vk (64th largest)
    }
    float sum = 0.0f;
    for (int j = l; j < SS; j += 32) {
      float v = row[j];
      float e = (v >= thr) ? __expf(v - mx) : 0.0f;
      row[j] = e;
      sum += e;
    }
    for (int msk = 16; msk; msk >>= 1) sum += __shfl_xor(sum, msk, 32);
    float inv = 1.0f / sum;
    for (int j = l; j < SS; j += 32) row[j] *= inv;
  }
  __syncthreads();

  // ---- Step 4: O(32x64) = P(32x2048) @ V(2048x64), causal-truncated.
  // V tiles (32x64 bf16) staged in LDS; TDM double-buffered when available.
  int rt = w >> 2, ct = w & 3;             // 2x4 tile grid over 8 waves
  v8f c = {};
  const float* aBase = sc + (size_t)(rt * 16 + lo16) * LDP;
  int nIter = rb + 1;                      // (qs+32)/32

#if HAVE_TDM
  if (w == 0) tdm_load_tile(0, Vh);        // preload tile 0 -> buf0
  for (int i = 0; i < nIter; ++i) {
    if (w == 0) __builtin_amdgcn_s_wait_tensorcnt(0);
    __syncthreads();                       // tile i visible to all waves
    if (w == 0 && i + 1 < nIter)
      tdm_load_tile(((unsigned)(i + 1) & 1u) * 4096u,
                    Vh + (size_t)(i + 1) * 32 * DKK);
    const unsigned short* vb = vls + (i & 1) * 2048;
    int k0 = i * 32;
    Frag16 fa, fb;
    int base = k0 + hi * 8;
#pragma unroll
    for (int jj = 0; jj < 8; ++jj) {
      fa.us[jj]     = f2bf(aBase[base + jj]);
      fa.us[8 + jj] = f2bf(aBase[base + 16 + jj]);
    }
#pragma unroll
    for (int jj = 0; jj < 16; ++jj)
      fb.us[jj] = vb[(hi * 8 + jj + (jj >= 8 ? 8 : 0)) * DKK + ct * 16 + lo16];
    c = __builtin_amdgcn_wmma_f32_16x16x32_bf16(false, fa.bf, false, fb.bf,
                                                (short)0, c, false, false);
    __syncthreads();                       // done reading buf before reuse
  }
#else
  for (int i = 0; i < nIter; ++i) {
    __syncthreads();
    for (int j = tid; j < 2048; j += 256)  // cooperative stage of tile i
      vls[j] = Vh[(size_t)i * 2048 + j];
    __syncthreads();
    int k0 = i * 32;
    Frag16 fa, fb;
    int base = k0 + hi * 8;
#pragma unroll
    for (int jj = 0; jj < 8; ++jj) {
      fa.us[jj]     = f2bf(aBase[base + jj]);
      fa.us[8 + jj] = f2bf(aBase[base + 16 + jj]);
    }
#pragma unroll
    for (int jj = 0; jj < 16; ++jj)
      fb.us[jj] = vls[(hi * 8 + jj + (jj >= 8 ? 8 : 0)) * DKK + ct * 16 + lo16];
    c = __builtin_amdgcn_wmma_f32_16x16x32_bf16(false, fa.bf, false, fb.bf,
                                                (short)0, c, false, false);
  }
#endif

  for (int r = 0; r < 8; ++r) {
    int mrow = rt * 16 + r + hi * 8;
    int sg   = qs + mrow;
    AO[((size_t)b * SS + sg) * DD + h * DKK + ct * 16 + lo16] = f2bf(c[r]);
  }
}

// ---------------------------------------------------------------- launch
extern "C" void kernel_launch(void* const* d_in, const int* in_sizes, int n_in,
                              void* d_out, int out_size, void* d_ws, size_t ws_size,
                              hipStream_t stream) {
  const float* q  = (const float*)d_in[0];
  const float* k  = (const float*)d_in[1];
  const float* v  = (const float*)d_in[2];
  const float* Wq = (const float*)d_in[3];
  const float* bq = (const float*)d_in[4];
  const float* Wk = (const float*)d_in[5];
  const float* bk = (const float*)d_in[6];
  const float* Wv = (const float*)d_in[7];
  const float* bv = (const float*)d_in[8];
  const float* Wo = (const float*)d_in[9];
  const float* bo = (const float*)d_in[10];
  (void)in_sizes; (void)n_in; (void)out_size; (void)ws_size;

  const size_t nTok = (size_t)BB * SS * DD;   // 4,194,304 elems
  const size_t nW   = (size_t)DD * DD;        // 1,048,576 elems

  char* ws = (char*)d_ws;
  unsigned short* XQ  = (unsigned short*)ws;  ws += nTok * 2;
  unsigned short* XK  = (unsigned short*)ws;  ws += nTok * 2;
  unsigned short* XV  = (unsigned short*)ws;  ws += nTok * 2;
  unsigned short* WQb = (unsigned short*)ws;  ws += nW * 2;
  unsigned short* WKb = (unsigned short*)ws;  ws += nW * 2;
  unsigned short* WVb = (unsigned short*)ws;  ws += nW * 2;
  unsigned short* WOb = (unsigned short*)ws;  ws += nW * 2;
  unsigned short* Qp  = (unsigned short*)ws;  ws += nTok * 2;
  unsigned short* Kp  = (unsigned short*)ws;  ws += nTok * 2;
  unsigned short* Vp  = (unsigned short*)ws;  ws += nTok * 2;
  unsigned short* AO  = (unsigned short*)ws;  ws += nTok * 2;   // ~64 MB total

  // fp32 -> bf16
  cvt_f32_bf16<<<2048, 256, 0, stream>>>(q,  XQ,  (int)nTok);
  cvt_f32_bf16<<<2048, 256, 0, stream>>>(k,  XK,  (int)nTok);
  cvt_f32_bf16<<<2048, 256, 0, stream>>>(v,  XV,  (int)nTok);
  cvt_f32_bf16<<<1024, 256, 0, stream>>>(Wq, WQb, (int)nW);
  cvt_f32_bf16<<<1024, 256, 0, stream>>>(Wk, WKb, (int)nW);
  cvt_f32_bf16<<<1024, 256, 0, stream>>>(Wv, WVb, (int)nW);
  cvt_f32_bf16<<<1024, 256, 0, stream>>>(Wo, WOb, (int)nW);

  // Q/K/V projections: 4096 16x64 strips, 8 waves/block -> 512 blocks
  proj_gemm<<<512, 256, 0, stream>>>(XQ, WQb, bq, Qp);
  proj_gemm<<<512, 256, 0, stream>>>(XK, WKb, bk, Kp);
  proj_gemm<<<512, 256, 0, stream>>>(XV, WVb, bv, Vp);

  // Attention: B*H*(S/32) = 2048 blocks, ~264.5 KB dynamic LDS
  topk_attn<<<2048, 256, ATTN_LDS, stream>>>(Qp, Kp, Vp, AO);

  // Output projection -> fp32
  out_gemm<<<512, 256, 0, stream>>>(AO, WOb, bo, (float*)d_out);
}
